// WindowAttention_48430051229767
// MI455X (gfx1250) — matrix operand order repaired and checked by reading, not compiled, not run on hardware
//
#include <hip/hip_runtime.h>

typedef __attribute__((ext_vector_type(16))) __bf16 v16bf;
typedef __attribute__((ext_vector_type(8)))  float  v8f;
typedef __attribute__((ext_vector_type(4)))  unsigned short u16x4;

// float -> bf16 via native conversion, raw u16 for storage
static __device__ __forceinline__ unsigned short f2bf_u(float f) {
    return __builtin_bit_cast(unsigned short, (__bf16)f);
}

// ---- WMMA fragment-native tile layout: [tile][lane 0..31][16 bf16] = 512 ush ----
// A (16x32 MxK): lane = m + 16*((kk>>3)&1), e = (kk&7) + 8*(kk>>4)
static __device__ __forceinline__ int a_off(int tile, int m, int kk) {
    return tile * 512 + ((((kk >> 3) & 1) << 4) + m) * 16 + ((kk & 7) + ((kk >> 4) << 3));
}
// B (32x16 KxN): lane = n + 16*((kk>>3)&1), e = (kk&7) + 8*(kk>>4)
static __device__ __forceinline__ int b_off(int tile, int kk, int n) {
    return tile * 512 + ((n + (((kk >> 3) & 1) << 4)) << 4) + ((kk & 7) + ((kk >> 4) << 3));
}
// Fragment load (works for LDS or global): 32 contiguous bytes/lane -> b128 pair
static __device__ __forceinline__ v16bf frag(const unsigned short* base, int tile, int lane) {
    return *(const v16bf*)(base + tile * 512 + lane * 16);
}

static __device__ __forceinline__ v8f wmma_bf16(v16bf a, v16bf b, v8f c) {
    return __builtin_amdgcn_wmma_f32_16x16x32_bf16(false, a, false, b,
                                                   (short)0, c, false, false);
}

// ---------------------------------------------------------------------------
// Prep: transform a KxN fp32 row-major weight matrix into bf16 B-fragment
// tiles: tile = (kkg>>5)*(N/16) + (n>>4). One-time cost per launch.
// Grid = (K) blocks x 128 threads.
// ---------------------------------------------------------------------------
__global__ __launch_bounds__(128) void wfrag_prep_kernel(
    const float* __restrict__ src, unsigned short* __restrict__ dst, int N) {
    const int kkg = blockIdx.x;                 // global K row
    const int ks = kkg >> 5, kk = kkg & 31;
    const int nct = N >> 4;
    const float* row = src + (size_t)kkg * N;
    for (int j4 = threadIdx.x * 4; j4 < N; j4 += blockDim.x * 4) {
        float4 v = *(const float4*)(row + j4);
        unsigned short* d = dst + b_off(ks * nct + (j4 >> 4), kk, j4 & 15);
        d[0]  = f2bf_u(v.x); d[16] = f2bf_u(v.y);
        d[32] = f2bf_u(v.z); d[48] = f2bf_u(v.w);
    }
}

// ---------------------------------------------------------------------------
// Kernel 1: per-(window, head) fused QKV projection + windowed attention.
// B fragments of W_qkv stream directly from global (pre-swizzled bf16).
// Attention output written to scratch already in A-fragment layout.
// Block = 128 threads (4 waves). Grid = (1024 windows, 12 heads).
// ---------------------------------------------------------------------------
__global__ __launch_bounds__(128) void wattn_qkv_attn_kernel(
    const float* __restrict__ x, const unsigned short* __restrict__ WqkvF,
    const float* __restrict__ bqkv, unsigned short* __restrict__ attnF) {
    __shared__ alignas(32) unsigned short Xf[48 * 512];  // A frags: tile = rowTile*12 + kTile
    __shared__ alignas(32) unsigned short Qf[4 * 512];   // A frags: tile = rowTile
    __shared__ alignas(32) unsigned short Kf[4 * 512];   // B(=K^T) frags: tile = key tile
    __shared__ alignas(32) unsigned short Vf[4 * 512];   // B frags: tile = ksb*2 + ct
    __shared__ alignas(32) unsigned short Pf[8 * 512];   // A frags: tile = rowTile*2 + kTile
    __shared__ float bias_s[96];

    const int gw   = blockIdx.x;           // b*64 + window
    const int h    = blockIdx.y;           // head
    const int b    = gw >> 6;
    const int win  = gw & 63;
    const int wy   = win >> 3, wx = win & 7;
    const int tid  = threadIdx.x;
    const int lane = tid & 31;
    const int wid  = tid >> 5;             // wave owns token rows [16*wid, 16*wid+16)

    // ---- Stage X window into A-fragment layout: 2 threads per token row ----
    {
        int t  = tid >> 1;                                  // token 0..63
        int m  = t & 15, rt = t >> 4;
        int n  = ((wy << 3) + (t >> 3)) * 64 + (wx << 3) + (t & 7);
        int cbh = (tid & 1) * 192;                          // half-row base
        const float* src = &x[((size_t)b * 4096 + n) * 384 + cbh];
#pragma unroll 8
        for (int jj = 0; jj < 48; ++jj) {
            int c4 = cbh + jj * 4;
            float4 v = *(const float4*)(src + jj * 4);
            u16x4 pk;
            pk[0] = f2bf_u(v.x); pk[1] = f2bf_u(v.y);
            pk[2] = f2bf_u(v.z); pk[3] = f2bf_u(v.w);
            *(u16x4*)(Xf + a_off(rt * 12 + (c4 >> 5), m, c4 & 31)) = pk;
        }
    }
    if (tid < 96) bias_s[tid] = bqkv[(tid >> 5) * 384 + h * 32 + (tid & 31)];
    __syncthreads();

    // ---- QKV GEMM: (64 x 384) @ (384 x 96); B frags direct from global ----
    // W_qkv col-tile for ct: mat = ct>>1 (Q/K/V), tile = mat*24 + 2h + (ct&1).
    v8f acc[6];
#pragma unroll
    for (int ct = 0; ct < 6; ++ct)
#pragma unroll
        for (int r = 0; r < 8; ++r) acc[ct][r] = 0.0f;

    for (int ks = 0; ks < 12; ++ks) {
        v16bf a = frag(Xf, wid * 12 + ks, lane);
        v16bf bf[6];
#pragma unroll
        for (int ct = 0; ct < 6; ++ct)
            bf[ct] = frag(WqkvF, ks * 72 + (ct >> 1) * 24 + (h << 1) + (ct & 1), lane);
#pragma unroll
        for (int ct = 0; ct < 6; ++ct) acc[ct] = wmma_bf16(a, bf[ct], acc[ct]);
    }

    // ---- bias + scatter into fragment-native Q / K^T / V ----
#pragma unroll
    for (int ct = 0; ct < 6; ++ct) {
        const int mat = ct >> 1;                        // 0=Q, 1=K, 2=V (compile-time)
        int jc = ((ct & 1) << 4) + (lane & 15);         // head-dim index 0..31
        float bv = bias_s[ct * 16 + (lane & 15)];
#pragma unroll
        for (int r = 0; r < 8; ++r) {
            int row = wid * 16 + ((lane >> 4) << 3) + r;   // token / key index
            unsigned short val = f2bf_u(acc[ct][r] + bv);
            if (mat == 0)      Qf[a_off(wid, row & 15, jc)] = val;
            else if (mat == 1) Kf[b_off(wid, jc, row & 15)] = val;
            else               Vf[b_off(((row >> 5) << 1) + (jc >> 4), row & 31, jc & 15)] = val;
        }
    }
    __syncthreads();

    // ---- S = (Q @ K^T) * scale ; head_dim 32 == one WMMA K-step ----
    const float scale = 0.17677669529663687f;  // 32^-0.5
    v8f st[4];
    {
        v16bf aq = frag(Qf, wid, lane);
        v16bf bk[4];
#pragma unroll
        for (int t = 0; t < 4; ++t) bk[t] = frag(Kf, t, lane);
#pragma unroll
        for (int t = 0; t < 4; ++t) {
            v8f z;
#pragma unroll
            for (int r = 0; r < 8; ++r) z[r] = 0.0f;
            st[t] = wmma_bf16(aq, bk[t], z);
        }
    }

    // ---- row softmax over 64 keys; P scattered into A-fragment layout ----
#pragma unroll
    for (int vr = 0; vr < 8; ++vr) {
        float m = st[0][vr] * scale;
#pragma unroll
        for (int t = 1; t < 4; ++t) m = fmaxf(m, st[t][vr] * scale);
#pragma unroll
        for (int msk = 1; msk < 16; msk <<= 1) m = fmaxf(m, __shfl_xor(m, msk, 32));
        float e[4], s = 0.0f;
#pragma unroll
        for (int t = 0; t < 4; ++t) { e[t] = __expf(st[t][vr] * scale - m); s += e[t]; }
#pragma unroll
        for (int msk = 1; msk < 16; msk <<= 1) s += __shfl_xor(s, msk, 32);
        float inv = 1.0f / s;
        int row = wid * 16 + ((lane >> 4) << 3) + vr;
#pragma unroll
        for (int t = 0; t < 4; ++t) {
            int col = t * 16 + (lane & 15);
            Pf[a_off((wid << 1) + (col >> 5), row & 15, col & 31)] = f2bf_u(e[t] * inv);
        }
    }
    __syncthreads();

    // ---- O = P @ V; store O directly as A-fragment tiles in scratch ----
    // attnF window slab = 48 tiles (rowTile*12 + kTile); head h == kTile h.
    v16bf ap0 = frag(Pf, (wid << 1) + 0, lane);
    v16bf ap1 = frag(Pf, (wid << 1) + 1, lane);
    v16bf bv0 = frag(Vf, 0, lane), bv1 = frag(Vf, 1, lane);
    v16bf bv2 = frag(Vf, 2, lane), bv3 = frag(Vf, 3, lane);
    unsigned short* otile = attnF + ((size_t)gw * 48 + wid * 12 + h) * 512;
#pragma unroll
    for (int ct = 0; ct < 2; ++ct) {
        v8f o;
#pragma unroll
        for (int r = 0; r < 8; ++r) o[r] = 0.0f;
        o = wmma_bf16(ap0, ct ? bv1 : bv0, o);   // keys  0..31
        o = wmma_bf16(ap1, ct ? bv3 : bv2, o);   // keys 32..63
#pragma unroll
        for (int r = 0; r < 8; ++r) {
            int m  = ((lane >> 4) << 3) + r;
            int kk = ct * 16 + (lane & 15);
            otile[a_off(0, m, kk)] = f2bf_u(o[r]);
        }
    }
}

// ---------------------------------------------------------------------------
// Kernel 2: projection (65536 x 384) @ (384 x 384). Fully LDS-free and
// barrier-free: A and B fragments stream from pre-swizzled global tiles.
// Epilogue scatters through the inverse window permutation into (B,N,C) fp32.
// Block = 128 threads (4 waves), 64x64 tile. Grid = (1024, 6).
// ---------------------------------------------------------------------------
__global__ __launch_bounds__(128) void wattn_proj_kernel(
    const unsigned short* __restrict__ attnF, const unsigned short* __restrict__ WprojF,
    float* __restrict__ out) {
    const int gw   = blockIdx.x;   // window == 64-row block
    const int cb   = blockIdx.y;   // 64-col block (0..5)
    const int lane = threadIdx.x & 31;
    const int wid  = threadIdx.x >> 5;

    v8f acc[4];
#pragma unroll
    for (int ct = 0; ct < 4; ++ct)
#pragma unroll
        for (int r = 0; r < 8; ++r) acc[ct][r] = 0.0f;

    const unsigned short* arow = attnF + ((size_t)gw * 48 + wid * 12) * 512;
#pragma unroll 2
    for (int ks = 0; ks < 12; ++ks) {
        v16bf a = frag(arow, ks, lane);
        v16bf bf[4];
#pragma unroll
        for (int ct = 0; ct < 4; ++ct)
            bf[ct] = frag(WprojF, ks * 24 + cb * 4 + ct, lane);
#pragma unroll
        for (int ct = 0; ct < 4; ++ct) acc[ct] = wmma_bf16(a, bf[ct], acc[ct]);
    }

    const int b = gw >> 6, win = gw & 63, wy = win >> 3, wx = win & 7;
#pragma unroll
    for (int ct = 0; ct < 4; ++ct) {
        int col = cb * 64 + ct * 16 + (lane & 15);
#pragma unroll
        for (int r = 0; r < 8; ++r) {
            int t = wid * 16 + ((lane >> 4) << 3) + r;
            int n = ((wy << 3) + (t >> 3)) * 64 + (wx << 3) + (t & 7);
            out[((size_t)b * 4096 + n) * 384 + col] = acc[ct][r];
        }
    }
}

extern "C" void kernel_launch(void* const* d_in, const int* in_sizes, int n_in,
                              void* d_out, int out_size, void* d_ws, size_t ws_size,
                              hipStream_t stream) {
    const float* x     = (const float*)d_in[0];
    const float* Wqkv  = (const float*)d_in[1];
    const float* bqkv  = (const float*)d_in[2];
    const float* Wproj = (const float*)d_in[3];
    float* out = (float*)d_out;

    // Scratch layout (halfword units):
    //   [0, 25165824)          attnF : 1024 windows * 48 tiles * 512   (48 MB)
    //   [25165824, 25608192)   WqkvF : 12*72 tiles * 512               (864 KB)
    //   [25608192, 25755648)   WprojF: 12*24 tiles * 512               (288 KB)
    unsigned short* attnF  = (unsigned short*)d_ws;
    unsigned short* WqkvF  = attnF + (size_t)25165824;
    unsigned short* WprojF = attnF + (size_t)25608192;

    wfrag_prep_kernel<<<384, 128, 0, stream>>>(Wqkv, WqkvF, 1152);
    wfrag_prep_kernel<<<384, 128, 0, stream>>>(Wproj, WprojF, 384);
    wattn_qkv_attn_kernel<<<dim3(1024, 12), 128, 0, stream>>>(x, WqkvF, bqkv, attnF);
    wattn_proj_kernel<<<dim3(1024, 6), 128, 0, stream>>>(attnF, WprojF, out);
}
